// LSTMRegressor_75462575390860
// MI455X (gfx1250) — compile-verified, matrix operand order checked by reading, not run on hardware
//
#include <hip/hip_runtime.h>
#include <hip/hip_fp16.h>

typedef __attribute__((ext_vector_type(16))) _Float16 v16h;
typedef __attribute__((ext_vector_type(4)))  _Float16 v4h;
typedef __attribute__((ext_vector_type(8)))  float    v8f;
typedef __attribute__((ext_vector_type(4)))  float    v4f;

#define HID    64
#define GATES  256   // 4*HID, PyTorch gate order i,f,g,o
#define BATCH  4096
#define SEQ    256
#define TILE_B 16

// Fast device math: approximate v_rcp_f32 (~1 ulp) and native v_tanh_f32.
__device__ __forceinline__ float fast_rcp(float x) {
    return __builtin_amdgcn_rcpf(x);
}
__device__ __forceinline__ float fast_sigmoid(float x) {
    return fast_rcp(1.0f + __expf(-x));   // v_exp + v_add + v_rcp
}
#if __has_builtin(__builtin_amdgcn_tanhf)
__device__ __forceinline__ float fast_tanh(float x) {
    return __builtin_amdgcn_tanhf(x);     // single v_tanh_f32
}
#else
__device__ __forceinline__ float fast_tanh(float x) {
    return 1.0f - 2.0f * fast_rcp(__expf(2.0f * x) + 1.0f);
}
#endif

// One LSTM layer. Each workgroup (256 threads = 8 wave32) owns TILE_B=16 batch
// rows for all SEQ timesteps. Weights live in VGPRs as WMMA B-fragments for
// the fused [W_ih | W_hh] matrix (K = KIN + 64). Per step:
//   g = [x_t | h_{t-1}] @ W^T          via v_wmma_f32_16x16x32_f16 (C starts at inline 0)
// then bias + sigmoid/tanh cell update. c stays in registers, h round-trips LDS.
template <int KIN, bool INF32>
__global__ __launch_bounds__(256)
void lstm_layer_kernel(const float* __restrict__ xin_f32,     // [B, SEQ, KIN] (layer 0)
                       const _Float16* xin_f16,               // [SEQ, B, HID] (layers 1/2)
                       const float* __restrict__ W_ih,        // [GATES, KIN]
                       const float* __restrict__ W_hh,        // [GATES, HID]
                       const float* __restrict__ b_ih,        // [GATES]
                       const float* __restrict__ b_hh,        // [GATES]
                       _Float16* hout)                        // [SEQ, B, HID]
{
    constexpr int K  = KIN + HID;   // 96 or 128
    constexpr int KT = K / 32;      // K-tiles of 32
    constexpr int KP = K + 8;       // padded LDS row stride; KP*2 is 16B-multiple

    __shared__ _Float16 inbuf[TILE_B * KP];    // [x_t | h_{t-1}] staging, f16
    __shared__ float    gbuf[TILE_B * GATES];  // pre-activation gates, f32

    const int tid  = threadIdx.x;
    const int lane = tid & 31;
    const int wave = tid >> 5;        // 0..7 -> gate columns [32*wave, 32*wave+32)
    const int hl   = lane >> 4;       // lane half (K-half selector in fragments)
    const int nl   = lane & 15;       // M (A) / N (B,C,D) index within tile
    const int b0   = blockIdx.x * TILE_B;

    // phase A/C ownership: thread owns row crow, hidden units [ch, ch+4)
    const int crow = tid >> 4;        // 0..15
    const int ch   = (tid & 15) * 4;  // 0,4,...,60

    // ---- prologue: weight B-fragments into registers ----
    // 16-bit fragment K-mapping per dword v (0..7): k = (v<4?0:16) + hl*8 + (v&3)*2
    v16h bfrag[2][KT];
    for (int nt = 0; nt < 2; ++nt) {
        const int n = wave * 32 + nt * 16 + nl;          // gate column
        for (int kt = 0; kt < KT; ++kt) {
            v16h f;
            for (int v = 0; v < 8; ++v) {
                const int k = kt * 32 + ((v < 4) ? 0 : 16) + hl * 8 + (v & 3) * 2;
                float w0, w1;
                if (k < KIN) { w0 = W_ih[n * KIN + k];         w1 = W_ih[n * KIN + k + 1]; }
                else         { w0 = W_hh[n * HID + (k - KIN)]; w1 = W_hh[n * HID + (k - KIN) + 1]; }
                f[2 * v]     = (_Float16)w0;
                f[2 * v + 1] = (_Float16)w1;
            }
            bfrag[nt][kt] = f;
        }
    }

    // Per-thread fused bias for phase C: [gate][jj] at column gate*64 + ch + jj
    v4f cbias[4];
    for (int g = 0; g < 4; ++g)
        for (int jj = 0; jj < 4; ++jj) {
            const int n = g * 64 + ch + jj;
            cbias[g][jj] = b_ih[n] + b_hh[n];
        }

    // h_{-1} = 0 (thread's own 4 cells)
    {
        v4h z;
        for (int jj = 0; jj < 4; ++jj) z[jj] = (_Float16)0.0f;
        *(v4h*)&inbuf[crow * KP + KIN + ch] = z;
    }
    float c_state[4] = {0.f, 0.f, 0.f, 0.f};

    for (int t = 0; t < SEQ; ++t) {
        // -- phase A: stage x_t (vectorized: one global load + one DS store) --
        if (INF32) {
            // layer 0: x is [B, SEQ, 32] f32; thread loads 2 floats
            const int col2 = (tid & 15) * 2;
            const float* p = &xin_f32[((size_t)(b0 + crow) * SEQ + t) * KIN + col2];
            if (col2 < KIN) {  // KIN=32: 16 lanes x 2 = 32 cols, all threads active
                inbuf[crow * KP + col2]     = (_Float16)p[0];
                inbuf[crow * KP + col2 + 1] = (_Float16)p[1];
            }
        } else {
            // layers 1/2: input is [SEQ, B, 64] f16; thread copies 4 halves (b64)
            const v4h hv = *(const v4h*)&xin_f16[((size_t)t * BATCH + b0 + crow) * HID + ch];
            *(v4h*)&inbuf[crow * KP + ch] = hv;
        }
        __syncthreads();  // x_t staged, h_{t-1} (from prev phase C) visible

        // -- phase B: WMMA gate GEMM --
        v16h a[KT];
#pragma unroll
        for (int kt = 0; kt < KT; ++kt) {
            v16h av;
#pragma unroll
            for (int v = 0; v < 8; ++v) {
                const int k = kt * 32 + ((v < 4) ? 0 : 16) + hl * 8 + (v & 3) * 2;
                av[2 * v]     = inbuf[nl * KP + k];
                av[2 * v + 1] = inbuf[nl * KP + k + 1];
            }
            a[kt] = av;
        }

        v8f acc0 = {};   // all-zero C folds to the inline-0 SRC2 operand
        v8f acc1 = {};
#pragma unroll
        for (int kt = 0; kt < KT; ++kt) {
            acc0 = __builtin_amdgcn_wmma_f32_16x16x32_f16(false, a[kt], false, bfrag[0][kt],
                                                          (short)0, acc0, false, false);
            acc1 = __builtin_amdgcn_wmma_f32_16x16x32_f16(false, a[kt], false, bfrag[1][kt],
                                                          (short)0, acc1, false, false);
        }
#pragma unroll
        for (int r = 0; r < 8; ++r) {
            const int m = r + 8 * hl;                     // C/D layout: M = vgpr + 8*(lane/16)
            gbuf[m * GATES + wave * 32 + nl]      = acc0[r];
            gbuf[m * GATES + wave * 32 + 16 + nl] = acc1[r];
        }
        __syncthreads();  // gates complete in LDS

        // -- phase C: cell update; thread owns (crow, ch..ch+3) --
        const v4f* gb = (const v4f*)&gbuf[crow * GATES];
        const v4f gi4 = gb[(0 * 64 + ch) >> 2];           // ds_load_b128 each
        const v4f gf4 = gb[(1 * 64 + ch) >> 2];
        const v4f gg4 = gb[(2 * 64 + ch) >> 2];
        const v4f go4 = gb[(3 * 64 + ch) >> 2];
        v4h hv;
#pragma unroll
        for (int jj = 0; jj < 4; ++jj) {
            const float gi = fast_sigmoid(gi4[jj] + cbias[0][jj]);
            const float gf = fast_sigmoid(gf4[jj] + cbias[1][jj]);
            const float gg = fast_tanh   (gg4[jj] + cbias[2][jj]);
            const float go = fast_sigmoid(go4[jj] + cbias[3][jj]);
            const float c  = gf * c_state[jj] + gi * gg;
            c_state[jj] = c;
            hv[jj] = (_Float16)(go * fast_tanh(c));
        }
        *(v4h*)&inbuf[crow * KP + KIN + ch] = hv;                              // b64 DS store
        *(v4h*)&hout[((size_t)t * BATCH + b0 + crow) * HID + ch] = hv;         // b64 global store
        // next iteration's __syncthreads orders these writes before the reads
    }
}

// Head: z = relu(h_last @ fc1_w^T + fc1_b); out = z @ fc2_w^T + fc2_b. Tiny.
__global__ __launch_bounds__(256)
void fc_head_kernel(const _Float16* __restrict__ hlast,   // [B, HID] (= ws + (SEQ-1)*B*HID)
                    const float* __restrict__ fc1_w,      // [32, HID]
                    const float* __restrict__ fc1_b,      // [32]
                    const float* __restrict__ fc2_w,      // [1, 32]
                    const float* __restrict__ fc2_b,      // [1]
                    float* __restrict__ out)              // [B, 1]
{
    __shared__ float w1[32 * HID];
    __shared__ float sb1[32];
    __shared__ float w2[32];
    const int tid = threadIdx.x;
    for (int i = tid; i < 32 * HID; i += 256) w1[i] = fc1_w[i];
    if (tid < 32) { sb1[tid] = fc1_b[tid]; w2[tid] = fc2_w[tid]; }
    __syncthreads();

    const int b = blockIdx.x * 256 + tid;
    float hreg[HID];
#pragma unroll
    for (int k = 0; k < HID; ++k) hreg[k] = (float)hlast[(size_t)b * HID + k];

    float acc2 = fc2_b[0];
#pragma unroll 4
    for (int j = 0; j < 32; ++j) {
        float s = sb1[j];
#pragma unroll
        for (int k = 0; k < HID; ++k) s = fmaf(hreg[k], w1[j * HID + k], s);
        s = fmaxf(s, 0.0f);
        acc2 = fmaf(s, w2[j], acc2);
    }
    out[b] = acc2;
}

extern "C" void kernel_launch(void* const* d_in, const int* in_sizes, int n_in,
                              void* d_out, int out_size, void* d_ws, size_t ws_size,
                              hipStream_t stream) {
    (void)in_sizes; (void)n_in; (void)out_size; (void)ws_size;
    const float* x     = (const float*)d_in[0];
    const float* W_ih0 = (const float*)d_in[1];
    const float* W_hh0 = (const float*)d_in[2];
    const float* b_ih0 = (const float*)d_in[3];
    const float* b_hh0 = (const float*)d_in[4];
    const float* W_ih1 = (const float*)d_in[5];
    const float* W_hh1 = (const float*)d_in[6];
    const float* b_ih1 = (const float*)d_in[7];
    const float* b_hh1 = (const float*)d_in[8];
    const float* W_ih2 = (const float*)d_in[9];
    const float* W_hh2 = (const float*)d_in[10];
    const float* b_ih2 = (const float*)d_in[11];
    const float* b_hh2 = (const float*)d_in[12];
    const float* fc1_w = (const float*)d_in[13];
    const float* fc1_b = (const float*)d_in[14];
    const float* fc2_w = (const float*)d_in[15];
    const float* fc2_b = (const float*)d_in[16];

    // Workspace: one f16 [SEQ, BATCH, HID] hidden-sequence buffer (128 MB).
    // Reused in place across layers (read h_prev[t] happens strictly before
    // write h_new[t] at the same address within each tile's step loop, and
    // fits in the 192 MB L2 so inter-layer traffic stays on-chip).
    _Float16* ws = (_Float16*)d_ws;

    const dim3 grid(BATCH / TILE_B);
    const dim3 block(256);
    lstm_layer_kernel<32, true ><<<grid, block, 0, stream>>>(x,       nullptr, W_ih0, W_hh0, b_ih0, b_hh0, ws);
    lstm_layer_kernel<64, false><<<grid, block, 0, stream>>>(nullptr, ws,      W_ih1, W_hh1, b_ih1, b_hh1, ws);
    lstm_layer_kernel<64, false><<<grid, block, 0, stream>>>(nullptr, ws,      W_ih2, W_hh2, b_ih2, b_hh2, ws);

    const _Float16* hlast = ws + (size_t)(SEQ - 1) * BATCH * HID;
    fc_head_kernel<<<dim3(BATCH / 256), block, 0, stream>>>(hlast, fc1_w, fc1_b, fc2_w, fc2_b, (float*)d_out);
}
